// DockPointNet_70325794504956
// MI455X (gfx1250) — matrix operand order repaired and checked by reading, not compiled, not run on hardware
//
#include <hip/hip_runtime.h>
#include <hip/hip_bf16.h>

// ---------------------------------------------------------------------------
// DockPointNet pipeline for gfx1250 (MI455X).
// Edge/row MLPs on the WMMA pipe (v_wmma_f32_16x16x32_f16). A-fragments are
// gathered with unconditional global_load_b128 runs (half*8 folded into the
// row pointer so all bounds checks fold at compile time), LayerNorm via
// intra-half shfl_xor, scatter via global_atomic_add_f32. Weights pre-packed
// to f16 B-fragment layout and staged through LDS.
// ---------------------------------------------------------------------------

typedef _Float16 half_t;
typedef half_t v16h __attribute__((ext_vector_type(16)));
typedef float  v8f  __attribute__((ext_vector_type(8)));

#define N_ATOMS   50000
#define N_RES     6250
#define C_IN      30
#define E_DIM     12
#define E_BOND    150000
#define E_RAD     800000
#define E_RES_RAD 400000
#define E_CORR    20000

// ---------------------------------------------------------------------------
// Utility kernels
// ---------------------------------------------------------------------------
__global__ void zero_kernel(float* p, int n) {
  int i = blockIdx.x * blockDim.x + threadIdx.x;
  if (i < n) p[i] = 0.0f;
}

// x [n,30] -> xpad [n,32], zero padded (row stride multiple of 16 floats)
__global__ void pad_x_kernel(const float* __restrict__ x, float* __restrict__ xpad, int n) {
  int i = blockIdx.x * blockDim.x + threadIdx.x;
  if (i >= n * 32) return;
  int row = i >> 5, c = i & 31;
  xpad[i] = (c < C_IN) ? x[row * C_IN + c] : 0.0f;
}

// Pack fp32 weight [K,N] into f16 WMMA B-fragment tiles.
// Tile (kt,nt) holds B 32x16 (KxN). Lane L, element e maps to
//   k = kt*32 + (e<8 ? 0 : 16) + (L<16 ? 0 : 8) + (e%8),  n = nt*16 + (L%16)
// Stored lane-major: tileBase + L*16 + e  (each lane loads 32B contiguous).
__global__ void pack_weight_kernel(const float* __restrict__ W, half_t* __restrict__ out,
                                   int K, int N, int Kpad) {
  int idx = blockIdx.x * blockDim.x + threadIdx.x;
  int ntiles = N / 16;
  int total = (Kpad / 32) * ntiles * 512;
  if (idx >= total) return;
  int tile = idx / 512;
  int rem  = idx % 512;
  int lane = rem / 16;
  int e    = rem % 16;
  int kt = tile / ntiles, nt = tile % ntiles;
  int k = kt * 32 + (e < 8 ? 0 : 16) + (lane < 16 ? 0 : 8) + (e & 7);
  int n = nt * 16 + (lane & 15);
  float v = (k < K) ? W[(size_t)k * N + n] : 0.0f;
  out[idx] = (half_t)v;
}

__global__ void deg_count_kernel(const int* __restrict__ dst, int nE, float* __restrict__ deg) {
  int i = blockIdx.x * blockDim.x + threadIdx.x;
  if (i < nE) atomicAdd(&deg[dst[i]], 1.0f);
}

__global__ void dinv_kernel(float* deg, int n) {
  int i = blockIdx.x * blockDim.x + threadIdx.x;
  if (i < n) deg[i] = rsqrtf(deg[i] + 1.0f);  // +1 self loop; always > 0
}

__global__ void pool_aux_kernel(const float* __restrict__ pos, const int* __restrict__ ridx,
                                float* __restrict__ cnt, float* __restrict__ possum, int n) {
  int i = blockIdx.x * blockDim.x + threadIdx.x;
  if (i >= n) return;
  int r = ridx[i];
  atomicAdd(&cnt[r], 1.0f);
  atomicAdd(&possum[r * 3 + 0], pos[i * 3 + 0]);
  atomicAdd(&possum[r * 3 + 1], pos[i * 3 + 1]);
  atomicAdd(&possum[r * 3 + 2], pos[i * 3 + 2]);
}

__global__ void res_pos_kernel(float* __restrict__ possum, const float* __restrict__ cnt, int n3) {
  int i = blockIdx.x * blockDim.x + threadIdx.x;
  if (i < n3) possum[i] /= fmaxf(cnt[i / 3], 1.0f);
}

// ---------------------------------------------------------------------------
// WMMA row-MLP kernel.
//   msg = [ xin[s] (+xin2[s]) , pos[s]-pos[d] if POSDIFF ]  (K padded to KPAD)
//   h   = LayerNorm(ReLU(msg @ W + b)) * g + beta
//   atomicAdd into acc[remap ? remap[d] : d]
// FINPAD: row stride of the input buffer (multiple of 16 floats; zero padded
//         beyond FINREAL). FINREAL: logical feature count (posdiff follows).
// One wave = one 16-row tile.
// ---------------------------------------------------------------------------
template<int FINPAD, int FINREAL, int KPAD, int NOUT, bool POSDIFF, bool HAS2>
__global__ void mlp_rows_kernel(
    const float* __restrict__ xin, const float* __restrict__ xin2,
    const float* __restrict__ pos,
    const int* __restrict__ src, const int* __restrict__ dst,
    int nedges, int nnodes,
    const int* __restrict__ remap,
    const half_t* __restrict__ Wp,
    const float* __restrict__ bias, const float* __restrict__ gamma,
    const float* __restrict__ beta,
    float* __restrict__ acc) {
  constexpr int KT = KPAD / 32;
  constexpr int NT = NOUT / 16;
  __shared__ __align__(64) half_t wlds[KT * NT * 512];
  {
    const uint4* s4 = (const uint4*)Wp;
    uint4* d4 = (uint4*)wlds;
    for (int i = threadIdx.x; i < KT * NT * 64; i += blockDim.x) d4[i] = s4[i];
  }
  __syncthreads();

  const int lane = threadIdx.x & 31;
  const int half = lane >> 4;
  const int r16  = lane & 15;
  const int nrows = nedges + nnodes;
  const int ntile = (nrows + 15) >> 4;
  int tile = blockIdx.x * (blockDim.x >> 5) + (threadIdx.x >> 5);
  if (tile >= ntile) return;  // uniform per wave: EXEC stays all-ones for WMMA

  int erow = tile * 16 + r16;
  int s = 0, d = 0;
  if (erow < nedges)      { s = src[erow]; d = dst[erow]; }
  else if (erow < nrows)  { s = d = erow - nedges; }

  // Fold half*8 into the row pointer: run offsets become compile-time consts.
  const float* xrh  = xin + (size_t)s * FINPAD + (half ? 8 : 0);
  const float* xrh2 = HAS2 ? (xin2 + (size_t)s * FINPAD + (half ? 8 : 0)) : nullptr;

  float pd0 = 0.f, pd1 = 0.f, pd2 = 0.f;
  if constexpr (POSDIFF) {
    pd0 = pos[s * 3 + 0] - pos[d * 3 + 0];
    pd1 = pos[s * 3 + 1] - pos[d * 3 + 1];
    pd2 = pos[s * 3 + 2] - pos[d * 3 + 2];
  }

  v8f c[NT];
#pragma unroll
  for (int nt = 0; nt < NT; ++nt) c[nt] = v8f{0.f,0.f,0.f,0.f,0.f,0.f,0.f,0.f};

#pragma unroll
  for (int kt = 0; kt < KT; ++kt) {
    float rr[2][8];
#pragma unroll
    for (int run = 0; run < 2; ++run) {
      const int cbase = kt * 32 + run * 16;   // k = cbase + j + half*8
      if (cbase + 16 <= FINPAD) {             // compile-time: run fully in-bounds
        float4 p0 = *(const float4*)(xrh + cbase);
        float4 p1 = *(const float4*)(xrh + cbase + 4);
        rr[run][0]=p0.x; rr[run][1]=p0.y; rr[run][2]=p0.z; rr[run][3]=p0.w;
        rr[run][4]=p1.x; rr[run][5]=p1.y; rr[run][6]=p1.z; rr[run][7]=p1.w;
        if constexpr (HAS2) {
          float4 q0 = *(const float4*)(xrh2 + cbase);
          float4 q1 = *(const float4*)(xrh2 + cbase + 4);
          rr[run][0]+=q0.x; rr[run][1]+=q0.y; rr[run][2]+=q0.z; rr[run][3]+=q0.w;
          rr[run][4]+=q1.x; rr[run][5]+=q1.y; rr[run][6]+=q1.z; rr[run][7]+=q1.w;
        }
      } else {
#pragma unroll
        for (int j = 0; j < 8; ++j) rr[run][j] = 0.f;
      }
      if constexpr (POSDIFF) {
        // k - FINREAL is t (half0) or t+8 (half1); t compile-time.
#pragma unroll
        for (int j = 0; j < 8; ++j) {
          const int t = cbase + j - FINREAL;
          if (t >= -8 && t < 3) {  // compile-time prune: element may need patching
            float v0 = (t < 0) ? rr[run][j] : (t == 0 ? pd0 : (t == 1 ? pd1 : pd2));
            const int u = t + 8;
            float v1 = (u < 0) ? rr[run][j]
                               : (u == 0 ? pd0 : (u == 1 ? pd1 : (u == 2 ? pd2 : 0.f)));
            rr[run][j] = half ? v1 : v0;
          }
        }
      }
    }
    v16h a;
#pragma unroll
    for (int e = 0; e < 8; ++e) a[e] = (half_t)rr[0][e];
#pragma unroll
    for (int e = 0; e < 8; ++e) a[e + 8] = (half_t)rr[1][e];
#pragma unroll
    for (int nt = 0; nt < NT; ++nt) {
      v16h b = *(const v16h*)(wlds + (kt * NT + nt) * 512 + lane * 16);
      c[nt] = __builtin_amdgcn_wmma_f32_16x16x32_f16(false, a, false, b,
                                                     (short)0, c[nt], false, false);
    }
  }

  float bcol[NT], gcol[NT], becol[NT];
#pragma unroll
  for (int nt = 0; nt < NT; ++nt) {
    int col = nt * 16 + r16;
    bcol[nt] = bias[col]; gcol[nt] = gamma[col]; becol[nt] = beta[col];
  }

  float h[NT][8];
#pragma unroll
  for (int nt = 0; nt < NT; ++nt)
#pragma unroll
    for (int r = 0; r < 8; ++r) {
      float v = c[nt][r] + bcol[nt];
      h[nt][r] = v > 0.f ? v : 0.f;
    }

  // LayerNorm per row M = half*8 + r; values live in 16 lanes x NT tiles
  float mu[8], rstd[8];
#pragma unroll
  for (int r = 0; r < 8; ++r) {
    float ps = 0.f, pss = 0.f;
#pragma unroll
    for (int nt = 0; nt < NT; ++nt) { float v = h[nt][r]; ps += v; pss += v * v; }
#pragma unroll
    for (int m = 1; m < 16; m <<= 1) {
      ps  += __shfl_xor(ps,  m, 32);
      pss += __shfl_xor(pss, m, 32);
    }
    float mean = ps * (1.0f / NOUT);
    float var  = pss * (1.0f / NOUT) - mean * mean;
    mu[r] = mean; rstd[r] = rsqrtf(var + 1e-5f);
  }

#pragma unroll
  for (int r = 0; r < 8; ++r) {
    int M = half * 8 + r;
    int dM = __shfl(d, M, 32);
    bool valid = (tile * 16 + M) < nrows;
    int tgt = (remap && valid) ? remap[dM] : dM;
#pragma unroll
    for (int nt = 0; nt < NT; ++nt) {
      float v = (h[nt][r] - mu[r]) * rstd[r] * gcol[nt] + becol[nt];
      if (valid) atomicAdd(acc + (size_t)tgt * NOUT + nt * 16 + r16, v);
    }
  }
}

// ---------------------------------------------------------------------------
// GCNEConv edge kernel: msg = ([x_s, e_attr] @ W) * dinv[s]*dinv[d] + bias,
// scatter-add. K = 42 (30 from xpad, 12 from e_attr) padded to 64, NOUT = 64.
// ---------------------------------------------------------------------------
__global__ void gcn_edge_kernel(
    const float* __restrict__ xpad, const float* __restrict__ ea,
    const int* __restrict__ src, const int* __restrict__ dst,
    int nedges, int nnodes,
    const half_t* __restrict__ Wp, const float* __restrict__ bias,
    const float* __restrict__ dinv, float* __restrict__ acc) {
  constexpr int KT = 2, NT = 4, NOUT = 64;
  __shared__ __align__(64) half_t wlds[KT * NT * 512];
  {
    const uint4* s4 = (const uint4*)Wp;
    uint4* d4 = (uint4*)wlds;
    for (int i = threadIdx.x; i < KT * NT * 64; i += blockDim.x) d4[i] = s4[i];
  }
  __syncthreads();

  const int lane = threadIdx.x & 31;
  const int half = lane >> 4;
  const int r16  = lane & 15;
  const int nrows = nedges + nnodes;
  const int ntile = (nrows + 15) >> 4;
  int tile = blockIdx.x * (blockDim.x >> 5) + (threadIdx.x >> 5);
  if (tile >= ntile) return;

  int erow = tile * 16 + r16;
  int s = 0, d = 0;
  bool realEdge = erow < nedges;
  if (realEdge)           { s = src[erow]; d = dst[erow]; }
  else if (erow < nrows)  { s = d = erow - nedges; }
  float nrm = dinv[s] * dinv[d];

  // edge attr: 12 floats, unconditional clamped load, zeroed for self loops
  float ear[12];
  {
    int ec = realEdge ? erow : 0;
    const float* er = ea + (size_t)ec * E_DIM;
    float4 e0 = *(const float4*)(er);
    float4 e1 = *(const float4*)(er + 4);
    float4 e2 = *(const float4*)(er + 8);
    ear[0]=e0.x; ear[1]=e0.y; ear[2]=e0.z; ear[3]=e0.w;
    ear[4]=e1.x; ear[5]=e1.y; ear[6]=e1.z; ear[7]=e1.w;
    ear[8]=e2.x; ear[9]=e2.y; ear[10]=e2.z; ear[11]=e2.w;
#pragma unroll
    for (int j = 0; j < 12; ++j) ear[j] = realEdge ? ear[j] : 0.f;
  }

  const float* xrh = xpad + (size_t)s * 32 + (half ? 8 : 0);

  v8f c[NT];
#pragma unroll
  for (int nt = 0; nt < NT; ++nt) c[nt] = v8f{0.f,0.f,0.f,0.f,0.f,0.f,0.f,0.f};

#pragma unroll
  for (int kt = 0; kt < KT; ++kt) {
    float rr[2][8];
#pragma unroll
    for (int run = 0; run < 2; ++run) {
      const int cbase = kt * 32 + run * 16;   // k = cbase + j + half*8
      if (cbase + 16 <= 32) {
        float4 p0 = *(const float4*)(xrh + cbase);
        float4 p1 = *(const float4*)(xrh + cbase + 4);
        rr[run][0]=p0.x; rr[run][1]=p0.y; rr[run][2]=p0.z; rr[run][3]=p0.w;
        rr[run][4]=p1.x; rr[run][5]=p1.y; rr[run][6]=p1.z; rr[run][7]=p1.w;
      } else {
#pragma unroll
        for (int j = 0; j < 8; ++j) rr[run][j] = 0.f;
      }
      // splice edge attr: u = k - 30; compile-time constant indices into ear[]
#pragma unroll
      for (int j = 0; j < 8; ++j) {
        const int t = cbase + j - C_IN;
        if (t >= -8 && t < E_DIM) {
          float v0 = (t < 0) ? rr[run][j] : ear[t < 0 ? 0 : t];
          const int u = t + 8;
          float v1 = (u < 0) ? rr[run][j] : (u < E_DIM ? ear[(u < 0 || u >= E_DIM) ? 0 : u] : 0.f);
          rr[run][j] = half ? v1 : v0;
        }
      }
    }
    v16h a;
#pragma unroll
    for (int e = 0; e < 8; ++e) a[e] = (half_t)rr[0][e];
#pragma unroll
    for (int e = 0; e < 8; ++e) a[e + 8] = (half_t)rr[1][e];
#pragma unroll
    for (int nt = 0; nt < NT; ++nt) {
      v16h b = *(const v16h*)(wlds + (kt * NT + nt) * 512 + lane * 16);
      c[nt] = __builtin_amdgcn_wmma_f32_16x16x32_f16(false, a, false, b,
                                                     (short)0, c[nt], false, false);
    }
  }

  float bcol[NT];
#pragma unroll
  for (int nt = 0; nt < NT; ++nt) bcol[nt] = bias[nt * 16 + r16];

#pragma unroll
  for (int r = 0; r < 8; ++r) {
    int M = half * 8 + r;
    int dM = __shfl(d, M, 32);
    float nM = __shfl(nrm, M, 32);
    bool valid = (tile * 16 + M) < nrows;
#pragma unroll
    for (int nt = 0; nt < NT; ++nt) {
      float v = c[nt][r] * nM + bcol[nt];
      if (valid) atomicAdd(acc + (size_t)dM * NOUT + nt * 16 + r16, v);
    }
  }
}

// ---------------------------------------------------------------------------
// Loss: CosineEmbeddingLoss with pos/neg reweighting over 20k pairs.
// ---------------------------------------------------------------------------
__global__ void loss_pair_kernel(const float* __restrict__ x3,
                                 const int* __restrict__ si, const int* __restrict__ ti,
                                 const int* __restrict__ y, float* __restrict__ accum,
                                 int npairs) {
  int i = blockIdx.x * blockDim.x + threadIdx.x;
  if (i >= npairs) return;
  const float4* a4 = (const float4*)(x3 + (size_t)si[i] * 128);
  const float4* b4 = (const float4*)(x3 + (size_t)ti[i] * 128);
  float dot = 0.f, na = 0.f, nb = 0.f;
#pragma unroll 4
  for (int j = 0; j < 32; ++j) {
    float4 av = a4[j], bv = b4[j];
    dot += av.x * bv.x + av.y * bv.y + av.z * bv.z + av.w * bv.w;
    na  += av.x * av.x + av.y * av.y + av.z * av.z + av.w * av.w;
    nb  += bv.x * bv.x + bv.y * bv.y + bv.z * bv.z + bv.w * bv.w;
  }
  float cs = dot / (fmaxf(sqrtf(na), 1e-8f) * fmaxf(sqrtf(nb), 1e-8f));
  if (y[i] == 1) {
    atomicAdd(&accum[0], 1.0f - cs);       // sum of pos unweighted
    atomicAdd(&accum[2], 1.0f);            // pos count
  } else {
    atomicAdd(&accum[1], fmaxf(cs, 0.0f)); // sum of neg unweighted
    atomicAdd(&accum[3], 1.0f);            // neg count
  }
}

__global__ void loss_final_kernel(const float* __restrict__ accum, float* __restrict__ out,
                                  int npairs) {
  if (threadIdx.x == 0 && blockIdx.x == 0) {
    float posw = accum[3] / accum[2];      // neg_count / pos_count
    out[0] = (accum[0] * posw + accum[1]) / (float)npairs;
  }
}

// ---------------------------------------------------------------------------
// Host launcher
// ---------------------------------------------------------------------------
extern "C" void kernel_launch(void* const* d_in, const int* in_sizes, int n_in,
                              void* d_out, int out_size, void* d_ws, size_t ws_size,
                              hipStream_t stream) {
  (void)in_sizes; (void)n_in; (void)out_size; (void)ws_size;

  const float* x      = (const float*)d_in[0];
  const float* pos    = (const float*)d_in[1];
  const float* eattr  = (const float*)d_in[2];
  const int*   ei     = (const int*)d_in[3];     // [2, E_BOND]
  const int*   rei    = (const int*)d_in[4];     // [2, E_RAD]
  const int*   rrei   = (const int*)d_in[5];     // [2, E_RES_RAD]
  const int*   ridx   = (const int*)d_in[6];
  const int*   sidx   = (const int*)d_in[7];
  const int*   tidx   = (const int*)d_in[8];
  const int*   ylab   = (const int*)d_in[9];
  const float* w_pc1  = (const float*)d_in[10];
  const float* b_pc1  = (const float*)d_in[11];
  const float* g_pc1  = (const float*)d_in[12];
  const float* be_pc1 = (const float*)d_in[13];
  const float* w_gcn  = (const float*)d_in[14];
  const float* b_gcn  = (const float*)d_in[15];
  const float* w_ae   = (const float*)d_in[16];
  const float* b_ae   = (const float*)d_in[17];
  const float* g_ae   = (const float*)d_in[18];
  const float* be_ae  = (const float*)d_in[19];
  const float* w_re   = (const float*)d_in[20];
  const float* b_re   = (const float*)d_in[21];
  const float* g_re   = (const float*)d_in[22];
  const float* be_re  = (const float*)d_in[23];
  const float* w_rc   = (const float*)d_in[24];
  const float* b_rc   = (const float*)d_in[25];
  const float* g_rc   = (const float*)d_in[26];
  const float* be_rc  = (const float*)d_in[27];
  const float* w_rg   = (const float*)d_in[28];
  const float* b_rg   = (const float*)d_in[29];
  const float* g_rg   = (const float*)d_in[30];
  const float* be_rg  = (const float*)d_in[31];

  // -------- workspace layout --------
  char* ws = (char*)d_ws;
  size_t off = 0;
  auto alloc = [&](size_t bytes) { size_t o = off; off = (off + bytes + 255) & ~(size_t)255; return o; };
  size_t o_wpc1 = alloc(2 * 4 * 512 * sizeof(half_t));   // K64  x N64
  size_t o_wgcn = alloc(2 * 4 * 512 * sizeof(half_t));   // K64  x N64
  size_t o_wae  = alloc(2 * 4 * 512 * sizeof(half_t));   // K64  x N64
  size_t o_wre  = alloc(2 * 4 * 512 * sizeof(half_t));   // K64  x N64
  size_t o_wrc  = alloc(3 * 8 * 512 * sizeof(half_t));   // K96  x N128
  size_t o_wrg  = alloc(4 * 8 * 512 * sizeof(half_t));   // K128 x N128
  size_t o_xpad = alloc((size_t)N_ATOMS * 32 * 4);       // padded x (fully rewritten)
  size_t zstart = off;
  size_t o_x1   = alloc((size_t)N_ATOMS * 64 * 4);
  size_t o_x2   = alloc((size_t)N_ATOMS * 64 * 4);
  size_t o_deg  = alloc((size_t)N_ATOMS * 4);
  size_t o_rxs  = alloc((size_t)N_RES * 64 * 4);
  size_t o_cnt  = alloc((size_t)N_RES * 4);
  size_t o_psum = alloc((size_t)N_RES * 3 * 4);
  size_t o_resx = alloc((size_t)N_RES * 64 * 4);
  size_t o_x3a  = alloc((size_t)N_RES * 128 * 4);
  size_t o_x3   = alloc((size_t)N_RES * 128 * 4);
  size_t o_lacc = alloc(4 * 4);
  size_t zend = off;

  half_t* Wpc1 = (half_t*)(ws + o_wpc1);
  half_t* Wgcn = (half_t*)(ws + o_wgcn);
  half_t* Wae  = (half_t*)(ws + o_wae);
  half_t* Wre  = (half_t*)(ws + o_wre);
  half_t* Wrc  = (half_t*)(ws + o_wrc);
  half_t* Wrg  = (half_t*)(ws + o_wrg);
  float* xpad  = (float*)(ws + o_xpad);
  float* x1acc = (float*)(ws + o_x1);
  float* x2acc = (float*)(ws + o_x2);
  float* dinv  = (float*)(ws + o_deg);
  float* rxsum = (float*)(ws + o_rxs);
  float* cnt   = (float*)(ws + o_cnt);
  float* psum  = (float*)(ws + o_psum);
  float* resx  = (float*)(ws + o_resx);
  float* x3acc = (float*)(ws + o_x3a);
  float* x3f   = (float*)(ws + o_x3);
  float* lacc  = (float*)(ws + o_lacc);

  const int TPB = 256;
  auto cdiv = [](long long a, long long b) { return (int)((a + b - 1) / b); };

  // 0) zero accumulators, pad x
  int nz = (int)((zend - zstart) / 4);
  zero_kernel<<<cdiv(nz, TPB), TPB, 0, stream>>>((float*)(ws + zstart), nz);
  pad_x_kernel<<<cdiv((long long)N_ATOMS * 32, TPB), TPB, 0, stream>>>(x, xpad, N_ATOMS);

  // 1) pack weights into f16 B-fragment tiles
  pack_weight_kernel<<<cdiv(2*4*512, TPB), TPB, 0, stream>>>(w_pc1, Wpc1, 33, 64, 64);
  pack_weight_kernel<<<cdiv(2*4*512, TPB), TPB, 0, stream>>>(w_gcn, Wgcn, 42, 64, 64);
  pack_weight_kernel<<<cdiv(2*4*512, TPB), TPB, 0, stream>>>(w_ae,  Wae,  64, 64, 64);
  pack_weight_kernel<<<cdiv(2*4*512, TPB), TPB, 0, stream>>>(w_re,  Wre,  64, 64, 64);
  pack_weight_kernel<<<cdiv(3*8*512, TPB), TPB, 0, stream>>>(w_rc,  Wrc,  67, 128, 96);
  pack_weight_kernel<<<cdiv(4*8*512, TPB), TPB, 0, stream>>>(w_rg,  Wrg, 128, 128, 128);

  // 2) GCN degree -> dinv (deg starts 0; +1 self loop folded into dinv kernel)
  deg_count_kernel<<<cdiv(E_BOND, TPB), TPB, 0, stream>>>(ei + E_BOND, E_BOND, dinv);
  dinv_kernel<<<cdiv(N_ATOMS, TPB), TPB, 0, stream>>>(dinv, N_ATOMS);

  // 3) PointConv1 over atom radius graph (+self loops) -> x1acc
  {
    int rows = E_RAD + N_ATOMS, tiles = cdiv(rows, 16);
    mlp_rows_kernel<32, 30, 64, 64, true, false><<<cdiv(tiles, TPB/32), TPB, 0, stream>>>(
        xpad, nullptr, pos, rei, rei + E_RAD, E_RAD, N_ATOMS, nullptr,
        Wpc1, b_pc1, g_pc1, be_pc1, x1acc);
  }

  // 4) GCNEConv over bond graph (+self loops) -> x2acc
  {
    int rows = E_BOND + N_ATOMS, tiles = cdiv(rows, 16);
    gcn_edge_kernel<<<cdiv(tiles, TPB/32), TPB, 0, stream>>>(
        xpad, eattr, ei, ei + E_BOND, E_BOND, N_ATOMS, Wgcn, b_gcn, dinv, x2acc);
  }

  // 5) atom_embed MLP on x1+x2, fused atom->residue sum pooling -> rxsum
  {
    int tiles = cdiv(N_ATOMS, 16);
    mlp_rows_kernel<64, 64, 64, 64, false, true><<<cdiv(tiles, TPB/32), TPB, 0, stream>>>(
        x1acc, x2acc, nullptr, nullptr, nullptr, 0, N_ATOMS, ridx,
        Wae, b_ae, g_ae, be_ae, rxsum);
  }

  // 6) residue counts + position means
  pool_aux_kernel<<<cdiv(N_ATOMS, TPB), TPB, 0, stream>>>(pos, ridx, cnt, psum, N_ATOMS);
  res_pos_kernel<<<cdiv(N_RES * 3, TPB), TPB, 0, stream>>>(psum, cnt, N_RES * 3);

  // 7) res_embed MLP -> resx  (atomicAdd into zeroed buffer == store)
  {
    int tiles = cdiv(N_RES, 16);
    mlp_rows_kernel<64, 64, 64, 64, false, false><<<cdiv(tiles, TPB/32), TPB, 0, stream>>>(
        rxsum, nullptr, nullptr, nullptr, nullptr, 0, N_RES, nullptr,
        Wre, b_re, g_re, be_re, resx);
  }

  // 8) PointConv2 over residue radius graph (+self loops) -> x3acc
  {
    int rows = E_RES_RAD + N_RES, tiles = cdiv(rows, 16);
    mlp_rows_kernel<64, 64, 96, 128, true, false><<<cdiv(tiles, TPB/32), TPB, 0, stream>>>(
        resx, nullptr, psum, rrei, rrei + E_RES_RAD, E_RES_RAD, N_RES, nullptr,
        Wrc, b_rc, g_rc, be_rc, x3acc);
  }

  // 9) global MLP 128->128 -> x3f
  {
    int tiles = cdiv(N_RES, 16);
    mlp_rows_kernel<128, 128, 128, 128, false, false><<<cdiv(tiles, TPB/32), TPB, 0, stream>>>(
        x3acc, nullptr, nullptr, nullptr, nullptr, 0, N_RES, nullptr,
        Wrg, b_rg, g_rg, be_rg, x3f);
  }

  // 10) cosine embedding loss
  loss_pair_kernel<<<cdiv(E_CORR, TPB), TPB, 0, stream>>>(x3f, sidx, tidx, ylab, lacc, E_CORR);
  loss_final_kernel<<<1, 32, 0, stream>>>(lacc, (float*)d_out, E_CORR);
}
